// Attention_11089605558565
// MI455X (gfx1250) — matrix operand order repaired and checked by reading, not compiled, not run on hardware
//
#include <hip/hip_runtime.h>
#include <hip/hip_bf16.h>

// Problem constants (match reference)
#define B_   32
#define TQ_  256
#define TK_  1024
#define SV_  1024
#define LV_  512
#define P_   256
#define H_   4
#define EPS_ 1e-5f

typedef __attribute__((ext_vector_type(16))) __bf16 bf16x16;
typedef __attribute__((ext_vector_type(8)))  float  f32x8;

__device__ __forceinline__ f32x8 wmma_bf16(bf16x16 a, bf16x16 b, f32x8 c) {
  // D = A(16x32 bf16) * B(32x16 bf16) + C(16x16 f32)
  return __builtin_amdgcn_wmma_f32_16x16x32_bf16(
      /*neg_a=*/false, a, /*neg_b=*/false, b,
      /*c_mod=*/(short)0, c, /*reuse_a=*/false, /*reuse_b=*/false);
}

// ---- fragment loaders (wave32 VGPR layouts per CDNA5 ISA 7.12.2) ----
// A 16x32 bf16 from fp32 row-major: lane m = lane&15; lanes>=16 shift K by +8/+24
__device__ __forceinline__ bf16x16 ldA_f32(const float* A, int lda, int row0, int k0, int lane) {
  const float* base = A + (size_t)(row0 + (lane & 15)) * lda + k0 + ((lane >> 4) << 3);
  bf16x16 r;
#pragma unroll
  for (int v = 0; v < 8; ++v) {
    int k = ((v & 4) << 2) + ((v & 3) << 1);  // v<4: 2v ; v>=4: 16+2(v-4)
    r[2 * v]     = (__bf16)base[k];
    r[2 * v + 1] = (__bf16)base[k + 1];
  }
  return r;
}
__device__ __forceinline__ bf16x16 ldA_bf16(const __bf16* A, int lda, int row0, int k0, int lane) {
  const __bf16* base = A + (size_t)(row0 + (lane & 15)) * lda + k0 + ((lane >> 4) << 3);
  bf16x16 r;
#pragma unroll
  for (int v = 0; v < 8; ++v) {
    int k = ((v & 4) << 2) + ((v & 3) << 1);
    r[2 * v]     = base[k];
    r[2 * v + 1] = base[k + 1];
  }
  return r;
}
// B 32x16 from row-major NxK bf16 (pre-transposed operand) -> lanes>=16 shift K by +8/+24 (same as A)
__device__ __forceinline__ bf16x16 ldB_nk_bf16(const __bf16* Bt, int ldb, int k0, int col0, int lane) {
  const __bf16* base = Bt + (size_t)(col0 + (lane & 15)) * ldb + k0 + ((lane >> 4) << 3);
  bf16x16 r;
#pragma unroll
  for (int v = 0; v < 8; ++v) { int k = ((v & 4) << 2) + ((v & 3) << 1); r[2 * v] = base[k]; r[2 * v + 1] = base[k + 1]; }
  return r;
}
// C/D 16x16 f32: VGPR v -> M = v + 8*(lane>=16), N = lane&15
__device__ __forceinline__ void stC_bf16(__bf16* C, int ldc, int row0, int col0, f32x8 acc, int lane) {
  __bf16* base = C + (size_t)(row0 + ((lane >> 4) << 3)) * ldc + col0 + (lane & 15);
#pragma unroll
  for (int v = 0; v < 8; ++v) base[(size_t)v * ldc] = (__bf16)acc[v];
}
__device__ __forceinline__ void stC_f32(float* C, int ldc, int row0, int col0, f32x8 acc, int lane) {
  float* base = C + (size_t)(row0 + ((lane >> 4) << 3)) * ldc + col0 + (lane & 15);
#pragma unroll
  for (int v = 0; v < 8; ++v) base[(size_t)v * ldc] = acc[v];
}

// order-preserving float<->uint for atomic max (scores can be negative)
__device__ __forceinline__ unsigned enc_f32(float f) {
  unsigned u = __float_as_uint(f);
  return (u & 0x80000000u) ? ~u : (u | 0x80000000u);
}
__device__ __forceinline__ float dec_f32(unsigned u) {
  unsigned b = (u & 0x80000000u) ? (u & 0x7FFFFFFFu) : ~u;
  return __uint_as_float(b);
}

#define Z8 {0.f,0.f,0.f,0.f,0.f,0.f,0.f,0.f}

// ---------------- kernels ----------------

__global__ void k_init(unsigned* head_max) {
  if (threadIdx.x < H_) head_max[threadIdx.x] = enc_f32(-__builtin_inff());
}

// Tiled transpose + downconvert: src fp32 [R,C] (ld=C) -> dst bf16 [C,R] (ld=R).
// 32x32 tile via LDS (padded stride 33 to dodge bank conflicts); coalesced both ways.
// grid(x = C/32, y = R/32, z = batches)
__global__ __launch_bounds__(256) void k_transpose(const float* __restrict__ src,
                                                   __bf16* __restrict__ dst,
                                                   int R, int C,
                                                   size_t sStride, size_t dStride) {
  src += (size_t)blockIdx.z * sStride;
  dst += (size_t)blockIdx.z * dStride;
  const int c0 = blockIdx.x * 32, r0 = blockIdx.y * 32;
  const int tx = threadIdx.x & 31, ty = threadIdx.x >> 5;   // 32 x 8
  __shared__ float t[32][33];
#pragma unroll
  for (int i = 0; i < 4; ++i)
    t[ty + 8 * i][tx] = src[(size_t)(r0 + ty + 8 * i) * C + c0 + tx];
  __syncthreads();
#pragma unroll
  for (int i = 0; i < 4; ++i)
    dst[(size_t)(c0 + ty + 8 * i) * R + r0 + tx] = (__bf16)t[tx][ty + 8 * i];
}

// m = s @ W_phi + b_phi  -> bf16 [8192,1024]; grid(x=16 Ntiles, y=64 Mtiles)
__global__ __launch_bounds__(256) void k_proj_m(const float* __restrict__ s,
                                                const __bf16* __restrict__ WphiT,  // [PH, SV]
                                                const float* __restrict__ bphi,
                                                __bf16* __restrict__ mout) {
  const int lane = threadIdx.x & 31, wv = threadIdx.x >> 5;
  const int row0 = blockIdx.y * 128 + (wv & 3) * 32;
  const int col0 = blockIdx.x * 64 + (wv >> 2) * 32;
  f32x8 a00 = Z8, a01 = Z8, a10 = Z8, a11 = Z8;
  for (int k0 = 0; k0 < SV_; k0 += 32) {
    if (k0 + 32 < SV_)  // next K slab -> global_prefetch_b8
      __builtin_prefetch(s + (size_t)(row0 + (lane & 15)) * SV_ + k0 + 32, 0, 1);
    bf16x16 fa0 = ldA_f32(s, SV_, row0, k0, lane);
    bf16x16 fa1 = ldA_f32(s, SV_, row0 + 16, k0, lane);
    bf16x16 fb0 = ldB_nk_bf16(WphiT, SV_, k0, col0, lane);
    bf16x16 fb1 = ldB_nk_bf16(WphiT, SV_, k0, col0 + 16, lane);
    a00 = wmma_bf16(fa0, fb0, a00); a01 = wmma_bf16(fa0, fb1, a01);
    a10 = wmma_bf16(fa1, fb0, a10); a11 = wmma_bf16(fa1, fb1, a11);
  }
  const float bias0 = bphi[col0 + (lane & 15)];
  const float bias1 = bphi[col0 + 16 + (lane & 15)];
#pragma unroll
  for (int v = 0; v < 8; ++v) { a00[v] += bias0; a10[v] += bias0; a01[v] += bias1; a11[v] += bias1; }
  stC_bf16(mout, H_ * P_, row0,      col0,      a00, lane);
  stC_bf16(mout, H_ * P_, row0,      col0 + 16, a01, lane);
  stC_bf16(mout, H_ * P_, row0 + 16, col0,      a10, lane);
  stC_bf16(mout, H_ * P_, row0 + 16, col0 + 16, a11, lane);
}

// n = h @ W_psi -> bf16 [32768,256]; grid(x=4, y=256)
__global__ __launch_bounds__(256) void k_proj_n(const float* __restrict__ h,
                                                const __bf16* __restrict__ WpsiT,  // [P, LV]
                                                __bf16* __restrict__ nout) {
  const int lane = threadIdx.x & 31, wv = threadIdx.x >> 5;
  const int row0 = blockIdx.y * 128 + (wv & 3) * 32;
  const int col0 = blockIdx.x * 64 + (wv >> 2) * 32;
  f32x8 a00 = Z8, a01 = Z8, a10 = Z8, a11 = Z8;
  for (int k0 = 0; k0 < LV_; k0 += 32) {
    if (k0 + 32 < LV_)
      __builtin_prefetch(h + (size_t)(row0 + (lane & 15)) * LV_ + k0 + 32, 0, 1);
    bf16x16 fa0 = ldA_f32(h, LV_, row0, k0, lane);
    bf16x16 fa1 = ldA_f32(h, LV_, row0 + 16, k0, lane);
    bf16x16 fb0 = ldB_nk_bf16(WpsiT, LV_, k0, col0, lane);
    bf16x16 fb1 = ldB_nk_bf16(WpsiT, LV_, k0, col0 + 16, lane);
    a00 = wmma_bf16(fa0, fb0, a00); a01 = wmma_bf16(fa0, fb1, a01);
    a10 = wmma_bf16(fa1, fb0, a10); a11 = wmma_bf16(fa1, fb1, a11);
  }
  stC_bf16(nout, P_, row0,      col0,      a00, lane);
  stC_bf16(nout, P_, row0,      col0 + 16, a01, lane);
  stC_bf16(nout, P_, row0 + 16, col0,      a10, lane);
  stC_bf16(nout, P_, row0 + 16, col0 + 16, a11, lane);
}

// e[b,h,q,k] = sum_p m[b,q,h*P+p] * n[b,k,p]; raw scores -> a_out slice of d_out,
// plus per-head global max (order-preserving atomicMax). grid(x = 2*16, y = H, z = B)
__global__ __launch_bounds__(256) void k_scores(const __bf16* __restrict__ m_bf,
                                                const __bf16* __restrict__ n_bf,
                                                float* __restrict__ e_out,
                                                unsigned* __restrict__ head_max) {
  const int lane = threadIdx.x & 31, wv = threadIdx.x >> 5, tid = threadIdx.x;
  const int b = blockIdx.z, hd = blockIdx.y;
  const int qt = blockIdx.x >> 4, kt = blockIdx.x & 15;
  const int row0 = qt * 128 + (wv & 3) * 32;   // within TQ
  const int col0 = kt * 64 + (wv >> 2) * 32;   // within TK
  const __bf16* A  = m_bf + (size_t)b * TQ_ * (H_ * P_) + hd * P_;  // [TQ, P] view, ld=H*P
  const __bf16* Bn = n_bf + (size_t)b * TK_ * P_;                   // [TK, P] (N-major)
  f32x8 a00 = Z8, a01 = Z8, a10 = Z8, a11 = Z8;
  for (int k0 = 0; k0 < P_; k0 += 32) {
    bf16x16 fa0 = ldA_bf16(A, H_ * P_, row0, k0, lane);
    bf16x16 fa1 = ldA_bf16(A, H_ * P_, row0 + 16, k0, lane);
    bf16x16 fb0 = ldB_nk_bf16(Bn, P_, k0, col0, lane);
    bf16x16 fb1 = ldB_nk_bf16(Bn, P_, k0, col0 + 16, lane);
    a00 = wmma_bf16(fa0, fb0, a00); a01 = wmma_bf16(fa0, fb1, a01);
    a10 = wmma_bf16(fa1, fb0, a10); a11 = wmma_bf16(fa1, fb1, a11);
  }
  float* E = e_out + (size_t)(b * H_ + hd) * TQ_ * TK_;
  stC_f32(E, TK_, row0,      col0,      a00, lane);
  stC_f32(E, TK_, row0,      col0 + 16, a01, lane);
  stC_f32(E, TK_, row0 + 16, col0,      a10, lane);
  stC_f32(E, TK_, row0 + 16, col0 + 16, a11, lane);
  // block max of raw scores (reference takes max BEFORE masking)
  float lm = -__builtin_inff();
#pragma unroll
  for (int v = 0; v < 8; ++v) {
    lm = fmaxf(lm, a00[v]); lm = fmaxf(lm, a01[v]);
    lm = fmaxf(lm, a10[v]); lm = fmaxf(lm, a11[v]);
  }
  __shared__ float sred[256];
  sred[tid] = lm; __syncthreads();
  for (int s2 = 128; s2 > 0; s2 >>= 1) {
    if (tid < s2) sred[tid] = fmaxf(sred[tid], sred[tid + s2]);
    __syncthreads();
  }
  if (tid == 0) atomicMax(head_max + hd, enc_f32(sred[0]));
}

// in-place masked softmax over k; one row (TK=1024) per 256-thread block
__global__ __launch_bounds__(256) void k_softmax(float* __restrict__ a_io,
                                                 const float* __restrict__ len_mask,
                                                 const unsigned* __restrict__ head_max) {
  const int q = blockIdx.x, hd = blockIdx.y, b = blockIdx.z, tid = threadIdx.x;
  const float emax = dec_f32(head_max[hd]);
  float* row = a_io + ((size_t)(b * H_ + hd) * TQ_ + q) * TK_;
  const float* mrow = len_mask + ((size_t)(b * TQ_ + q)) * TK_;
  float vals[TK_ / 256];
  float lsum = 0.f;
#pragma unroll
  for (int i = 0; i < TK_ / 256; ++i) {
    int k = tid + i * 256;
    float ev = __expf(row[k] - emax) * mrow[k];
    vals[i] = ev; lsum += ev;
  }
  __shared__ float sred[256];
  sred[tid] = lsum; __syncthreads();
  for (int s2 = 128; s2 > 0; s2 >>= 1) {
    if (tid < s2) sred[tid] += sred[tid + s2];
    __syncthreads();
  }
  const float inv = 1.f / (sred[0] + EPS_);
#pragma unroll
  for (int i = 0; i < TK_ / 256; ++i) row[tid + i * 256] = vals[i] * inv;
}

// ctx[b,q,h*LV+d] = sum_k a[b,h,q,k] * h[b,k,d]; B side from pre-transposed hT bf16.
// grid(x = 2*8, y = H, z = B)
__global__ __launch_bounds__(256) void k_ctx(const float* __restrict__ a_w,
                                             const __bf16* __restrict__ hT,   // [B][LV][TK]
                                             __bf16* __restrict__ ctx) {
  const int lane = threadIdx.x & 31, wv = threadIdx.x >> 5;
  const int b = blockIdx.z, hd = blockIdx.y;
  const int qt = blockIdx.x >> 3, dt = blockIdx.x & 7;
  const int row0 = qt * 128 + (wv & 3) * 32;   // within TQ
  const int col0 = dt * 64 + (wv >> 2) * 32;   // within LV
  const float*  A  = a_w + (size_t)(b * H_ + hd) * TQ_ * TK_;  // [TQ, TK]
  const __bf16* Bt = hT + (size_t)b * LV_ * TK_;               // [LV, TK] (N-major)
  f32x8 a00 = Z8, a01 = Z8, a10 = Z8, a11 = Z8;
  for (int k0 = 0; k0 < TK_; k0 += 32) {
    if (k0 + 32 < TK_)
      __builtin_prefetch(A + (size_t)(row0 + (lane & 15)) * TK_ + k0 + 32, 0, 1);
    bf16x16 fa0 = ldA_f32(A, TK_, row0, k0, lane);
    bf16x16 fa1 = ldA_f32(A, TK_, row0 + 16, k0, lane);
    bf16x16 fb0 = ldB_nk_bf16(Bt, TK_, k0, col0, lane);
    bf16x16 fb1 = ldB_nk_bf16(Bt, TK_, k0, col0 + 16, lane);
    a00 = wmma_bf16(fa0, fb0, a00); a01 = wmma_bf16(fa0, fb1, a01);
    a10 = wmma_bf16(fa1, fb0, a10); a11 = wmma_bf16(fa1, fb1, a11);
  }
  __bf16* C = ctx + (size_t)b * TQ_ * (H_ * LV_) + hd * LV_;
  stC_bf16(C, H_ * LV_, row0,      col0,      a00, lane);
  stC_bf16(C, H_ * LV_, row0,      col0 + 16, a01, lane);
  stC_bf16(C, H_ * LV_, row0 + 16, col0,      a10, lane);
  stC_bf16(C, H_ * LV_, row0 + 16, col0 + 16, a11, lane);
}

// c = ctx @ W_red + b_red -> f32 [8192,512]; grid(x = 8, y = 64)
__global__ __launch_bounds__(256) void k_red(const __bf16* __restrict__ ctx,
                                             const __bf16* __restrict__ WredT,  // [LV, H*LV]
                                             const float* __restrict__ bred,
                                             float* __restrict__ cout) {
  const int lane = threadIdx.x & 31, wv = threadIdx.x >> 5;
  const int row0 = blockIdx.y * 128 + (wv & 3) * 32;
  const int col0 = blockIdx.x * 64 + (wv >> 2) * 32;
  f32x8 a00 = Z8, a01 = Z8, a10 = Z8, a11 = Z8;
  const int K = H_ * LV_;
  for (int k0 = 0; k0 < K; k0 += 32) {
    bf16x16 fa0 = ldA_bf16(ctx, K, row0, k0, lane);
    bf16x16 fa1 = ldA_bf16(ctx, K, row0 + 16, k0, lane);
    bf16x16 fb0 = ldB_nk_bf16(WredT, K, k0, col0, lane);
    bf16x16 fb1 = ldB_nk_bf16(WredT, K, k0, col0 + 16, lane);
    a00 = wmma_bf16(fa0, fb0, a00); a01 = wmma_bf16(fa0, fb1, a01);
    a10 = wmma_bf16(fa1, fb0, a10); a11 = wmma_bf16(fa1, fb1, a11);
  }
  const float bias0 = bred[col0 + (lane & 15)];
  const float bias1 = bred[col0 + 16 + (lane & 15)];
#pragma unroll
  for (int v = 0; v < 8; ++v) { a00[v] += bias0; a10[v] += bias0; a01[v] += bias1; a11[v] += bias1; }
  stC_f32(cout, LV_, row0,      col0,      a00, lane);
  stC_f32(cout, LV_, row0,      col0 + 16, a01, lane);
  stC_f32(cout, LV_, row0 + 16, col0,      a10, lane);
  stC_f32(cout, LV_, row0 + 16, col0 + 16, a11, lane);
}

extern "C" void kernel_launch(void* const* d_in, const int* in_sizes, int n_in,
                              void* d_out, int out_size, void* d_ws, size_t ws_size,
                              hipStream_t stream) {
  const float* s        = (const float*)d_in[0];
  const float* h        = (const float*)d_in[1];
  const float* len_mask = (const float*)d_in[2];
  const float* W_phi    = (const float*)d_in[3];
  const float* b_phi    = (const float*)d_in[4];
  const float* W_psi    = (const float*)d_in[5];
  const float* W_red    = (const float*)d_in[6];
  const float* b_red    = (const float*)d_in[7];

  float* out_c = (float*)d_out;                       // [B,TQ,LV]
  float* out_a = out_c + (size_t)B_ * TQ_ * LV_;      // [B,H,TQ,TK]

  char* ws = (char*)d_ws;
  // [0,16M) m_bf, [16M,32M) n_bf  -- dead after k_scores, aliased by ctx_bf [0,32M)
  __bf16*   m_bf     = (__bf16*)(ws);
  __bf16*   n_bf     = (__bf16*)(ws + (16ull << 20));
  __bf16*   ctx_bf   = (__bf16*)(ws);                   // aliases m/n region
  __bf16*   hT       = (__bf16*)(ws + (32ull << 20));   // 32 MiB  [B][LV][TK]
  __bf16*   WphiT    = (__bf16*)(ws + (64ull << 20));   // 2 MiB   [PH, SV]
  __bf16*   WredT    = (__bf16*)(ws + (66ull << 20));   // 2 MiB   [LV, H*LV]
  __bf16*   WpsiT    = (__bf16*)(ws + (68ull << 20));   // 256 KiB [P, LV]
  unsigned* head_max = (unsigned*)(ws + (69ull << 20)); // 16 B

  k_init<<<1, 32, 0, stream>>>(head_max);
  // one-shot operand packing: fp32 [R,C] -> bf16 [C,R]
  k_transpose<<<dim3((H_*P_)/32, SV_/32, 1), 256, 0, stream>>>(W_phi, WphiT, SV_, H_*P_, 0, 0);
  k_transpose<<<dim3(P_/32, LV_/32, 1), 256, 0, stream>>>(W_psi, WpsiT, LV_, P_, 0, 0);
  k_transpose<<<dim3(LV_/32, (H_*LV_)/32, 1), 256, 0, stream>>>(W_red, WredT, H_*LV_, LV_, 0, 0);
  k_transpose<<<dim3(LV_/32, TK_/32, B_), 256, 0, stream>>>(h, hT, TK_, LV_,
                                                            (size_t)TK_*LV_, (size_t)LV_*TK_);

  k_proj_m<<<dim3(16, 64), 256, 0, stream>>>(s, WphiT, b_phi, m_bf);
  k_proj_n<<<dim3(4, 256), 256, 0, stream>>>(h, WpsiT, n_bf);
  k_scores<<<dim3(32, H_, B_), 256, 0, stream>>>(m_bf, n_bf, out_a, head_max);
  k_softmax<<<dim3(TQ_, H_, B_), 256, 0, stream>>>(out_a, len_mask, head_max);
  k_ctx<<<dim3(16, H_, B_), 256, 0, stream>>>(out_a, hT, ctx_bf);
  k_red<<<dim3(8, 64), 256, 0, stream>>>(ctx_bf, W_red ? WredT : WredT, b_red, out_c);
}
